// Model_32684701123317
// MI455X (gfx1250) — matrix operand order repaired and checked by reading, not compiled, not run on hardware
//
#include <hip/hip_runtime.h>
#include <stdint.h>

// ---------------------------------------------------------------------------
// Types for CDNA5 WMMA (gfx1250, wave32)
// ---------------------------------------------------------------------------
typedef __bf16        bf16x16 __attribute__((ext_vector_type(16)));
typedef float         f32x8   __attribute__((ext_vector_type(8)));
typedef unsigned int  u32x4   __attribute__((ext_vector_type(4)));

__device__ __forceinline__ unsigned short f2bf(float f) {
  union { float f; unsigned int u; } v; v.f = f;
  unsigned int u = v.u;
  if ((u & 0x7F800000u) != 0x7F800000u)       // RNE, pass through inf/nan
    u += 0x7FFFu + ((u >> 16) & 1u);
  return (unsigned short)(u >> 16);
}

// ---------------------------------------------------------------------------
// Generic bf16 WMMA GEMM:  C[M,N] = leaky?(scale * A[M,K] @ Bt[N,K]^T + bias)
// A row-major bf16, B stored transposed [N,K] bf16, C f32.
// Block = 256 thr = 8 waves (2 M-waves x 4 N-waves), tile 64(M) x 128(N).
// Each wave: 32x32 tile = 2x2 v_wmma_f32_16x16x32_bf16 per K-step.
// Requires M%64==0, N%128==0, K%32==0 (true for all call sites here).
// ---------------------------------------------------------------------------
__global__ __launch_bounds__(256)
void gemm_bf16_wmma(const unsigned short* __restrict__ A,
                    const unsigned short* __restrict__ Bt,
                    float* __restrict__ C,
                    int M, int N, int K,
                    const float* __restrict__ bias,
                    float scale, int leaky)
{
  const int lane = threadIdx.x & 31;
  const int wave = threadIdx.x >> 5;
  const int r    = lane & 15;
  const int hi   = lane >> 4;
  const int wm   = wave & 1;   // 0..1
  const int wn   = wave >> 1;  // 0..3
  const int m0   = blockIdx.y * 64  + wm * 32;
  const int n0   = blockIdx.x * 128 + wn * 32;

  f32x8 acc00 = {}, acc01 = {}, acc10 = {}, acc11 = {};

  // A fragment base per ISA layout: lane holds row (m+r), K = k0 + 8*hi + {0..7} and +16
  const unsigned short* a0  = A  + (size_t)(m0 + r)      * K + (hi << 3);
  const unsigned short* a1  = A  + (size_t)(m0 + 16 + r) * K + (hi << 3);
  // B fragment: lane holds col (n+r), K = k0 + 8*hi + {0..7} and +16
  const unsigned short* bp0 = Bt + (size_t)(n0 + r)      * K + (hi << 3);
  const unsigned short* bp1 = Bt + (size_t)(n0 + 16 + r) * K + (hi << 3);

  union FragU { bf16x16 v; u32x4 u[2]; };

  for (int k0 = 0; k0 < K; k0 += 32) {
    FragU fa0, fa1, fb0, fb1;
    fa0.u[0] = *reinterpret_cast<const u32x4*>(a0  + k0);
    fa0.u[1] = *reinterpret_cast<const u32x4*>(a0  + k0 + 16);
    fa1.u[0] = *reinterpret_cast<const u32x4*>(a1  + k0);
    fa1.u[1] = *reinterpret_cast<const u32x4*>(a1  + k0 + 16);
    fb0.u[0] = *reinterpret_cast<const u32x4*>(bp0 + k0);
    fb0.u[1] = *reinterpret_cast<const u32x4*>(bp0 + k0 + 16);
    fb1.u[0] = *reinterpret_cast<const u32x4*>(bp1 + k0);
    fb1.u[1] = *reinterpret_cast<const u32x4*>(bp1 + k0 + 16);

    acc00 = __builtin_amdgcn_wmma_f32_16x16x32_bf16(false, fa0.v, false, fb0.v,
                                                    (short)0, acc00, false, false);
    acc01 = __builtin_amdgcn_wmma_f32_16x16x32_bf16(false, fa0.v, false, fb1.v,
                                                    (short)0, acc01, false, false);
    acc10 = __builtin_amdgcn_wmma_f32_16x16x32_bf16(false, fa1.v, false, fb0.v,
                                                    (short)0, acc10, false, false);
    acc11 = __builtin_amdgcn_wmma_f32_16x16x32_bf16(false, fa1.v, false, fb1.v,
                                                    (short)0, acc11, false, false);
  }

  // C/D layout: VGPR e -> row m0+16*mi + 8*hi + e, col n0+16*ni + r
  auto store = [&](f32x8 accv, int mi, int ni) {
    union { f32x8 v; float f[8]; } res; res.v = accv;
    const int col = n0 + ni * 16 + r;
    const float badd = bias ? bias[col] : 0.0f;
    #pragma unroll
    for (int e = 0; e < 8; ++e) {
      const int row = m0 + mi * 16 + hi * 8 + e;
      float val = res.f[e] * scale + badd;
      if (leaky) val = val > 0.0f ? val : 0.1f * val;
      C[(size_t)row * N + col] = val;
    }
  };
  store(acc00, 0, 0); store(acc01, 0, 1);
  store(acc10, 1, 0); store(acc11, 1, 1);
}

// ---------------------------------------------------------------------------
// Direct conv1d + bias + LeakyReLU(0.1)  (layers 1-4: small Cin*K, VALU-bound)
// x:[N,Cin,Lin]  w:[Cout,Cin,K]  y:[N,Cout,Lout]
// ---------------------------------------------------------------------------
__global__ __launch_bounds__(256)
void conv1d_leaky(const float* __restrict__ x, const float* __restrict__ w,
                  const float* __restrict__ b, float* __restrict__ y,
                  int Nb, int Cin, int Lin, int Cout, int Lout,
                  int Kw, int stride, int pad)
{
  const int idx = blockIdx.x * 256 + threadIdx.x;
  const int total = Nb * Cout * Lout;
  if (idx >= total) return;
  const int l  = idx % Lout;
  const int t  = idx / Lout;
  const int co = t % Cout;
  const int n  = t / Cout;
  const float* xn = x + (size_t)n  * Cin * Lin;
  const float* wc = w + (size_t)co * Cin * Kw;
  const int base = l * stride - pad;
  float acc = b[co];
  for (int ci = 0; ci < Cin; ++ci) {
    const float* xr = xn + (size_t)ci * Lin;
    const float* wr = wc + (size_t)ci * Kw;
    #pragma unroll 8
    for (int k = 0; k < Kw; ++k) {
      const int li = base + k;
      if (li >= 0 && li < Lin) acc += xr[li] * wr[k];
    }
  }
  y[idx] = acc > 0.0f ? acc : 0.1f * acc;
}

// BatchNorm stats over (N, L) per channel: one block per channel
__global__ __launch_bounds__(256)
void bn_stats(const float* __restrict__ y, float* __restrict__ mu,
              float* __restrict__ rs, int Nb, int C, int L)
{
  const int c = blockIdx.x;
  __shared__ float s1[256], s2[256];
  float sum = 0.f, sq = 0.f;
  const int total = Nb * L;
  for (int i = threadIdx.x; i < total; i += 256) {
    const int n = i / L, l = i - n * L;
    const float v = y[((size_t)n * C + c) * L + l];
    sum += v; sq += v * v;
  }
  s1[threadIdx.x] = sum; s2[threadIdx.x] = sq;
  __syncthreads();
  for (int s = 128; s > 0; s >>= 1) {
    if (threadIdx.x < s) {
      s1[threadIdx.x] += s1[threadIdx.x + s];
      s2[threadIdx.x] += s2[threadIdx.x + s];
    }
    __syncthreads();
  }
  if (threadIdx.x == 0) {
    const float m   = s1[0] / (float)total;
    const float var = s2[0] / (float)total - m * m;  // biased var (matches jnp.var)
    mu[c] = m;
    rs[c] = rsqrtf(var + 1e-5f);
  }
}

__global__ __launch_bounds__(256)
void bn_norm(float* __restrict__ y, const float* __restrict__ mu,
             const float* __restrict__ rs, const float* __restrict__ g,
             const float* __restrict__ be, int Nb, int C, int L)
{
  const size_t idx = (size_t)blockIdx.x * 256 + threadIdx.x;
  const size_t total = (size_t)Nb * C * L;
  if (idx >= total) return;
  const int c = (int)((idx / L) % C);
  y[idx] = (y[idx] - mu[c]) * rs[c] * g[c] + be[c];
}

// Embedding gather: out[row*256 + e] = emb[tok[row]*256 + e]
__global__ __launch_bounds__(256)
void gather_embed(const int* __restrict__ tok, const float* __restrict__ emb,
                  float* __restrict__ out, int rows)
{
  const size_t idx = (size_t)blockIdx.x * 256 + threadIdx.x;
  if (idx >= (size_t)rows * 256) return;
  const int row = (int)(idx >> 8);
  const int e   = (int)(idx & 255);
  out[idx] = emb[(size_t)tok[row] * 256 + e];
}

__global__ __launch_bounds__(256)
void cvt_bf16(const float* __restrict__ in, unsigned short* __restrict__ out, size_t n)
{
  const size_t idx = (size_t)blockIdx.x * 256 + threadIdx.x;
  if (idx >= n) return;
  out[idx] = f2bf(in[idx]);
}

// in: [R, Cc] f32 -> out: [Cc, R] bf16
__global__ __launch_bounds__(256)
void transpose_cvt_bf16(const float* __restrict__ in, unsigned short* __restrict__ out,
                        int R, int Cc)
{
  const size_t idx = (size_t)blockIdx.x * 256 + threadIdx.x;
  if (idx >= (size_t)R * Cc) return;
  const int rr = (int)(idx / Cc);
  const int cc = (int)(idx - (size_t)rr * Cc);
  out[(size_t)cc * R + rr] = f2bf(in[idx]);
}

// Row softmax over 4096 cols, bf16 output (A operand of next GEMM)
__global__ __launch_bounds__(256)
void softmax4096_bf16(const float* __restrict__ in, unsigned short* __restrict__ out)
{
  const int row = blockIdx.x;
  const float* rp = in + (size_t)row * 4096;
  __shared__ float red[256];
  float v[16];
  float mx = -3.4e38f;
  #pragma unroll
  for (int i = 0; i < 16; ++i) { v[i] = rp[threadIdx.x + (i << 8)]; mx = fmaxf(mx, v[i]); }
  red[threadIdx.x] = mx; __syncthreads();
  for (int s = 128; s > 0; s >>= 1) {
    if (threadIdx.x < s) red[threadIdx.x] = fmaxf(red[threadIdx.x], red[threadIdx.x + s]);
    __syncthreads();
  }
  mx = red[0]; __syncthreads();
  float sum = 0.f;
  #pragma unroll
  for (int i = 0; i < 16; ++i) { v[i] = __expf(v[i] - mx); sum += v[i]; }
  red[threadIdx.x] = sum; __syncthreads();
  for (int s = 128; s > 0; s >>= 1) {
    if (threadIdx.x < s) red[threadIdx.x] += red[threadIdx.x + s];
    __syncthreads();
  }
  const float inv = 1.0f / red[0];
  unsigned short* op = out + (size_t)row * 4096;
  #pragma unroll
  for (int i = 0; i < 16; ++i) op[threadIdx.x + (i << 8)] = f2bf(v[i] * inv);
}

// In-place row softmax, arbitrary cols (vocab = 32000)
__global__ __launch_bounds__(256)
void softmax_inplace(float* __restrict__ x, int cols)
{
  const int row = blockIdx.x;
  float* rp = x + (size_t)row * cols;
  __shared__ float red[256];
  float mx = -3.4e38f;
  for (int j = threadIdx.x; j < cols; j += 256) mx = fmaxf(mx, rp[j]);
  red[threadIdx.x] = mx; __syncthreads();
  for (int s = 128; s > 0; s >>= 1) {
    if (threadIdx.x < s) red[threadIdx.x] = fmaxf(red[threadIdx.x], red[threadIdx.x + s]);
    __syncthreads();
  }
  mx = red[0]; __syncthreads();
  float sum = 0.f;
  for (int j = threadIdx.x; j < cols; j += 256) sum += __expf(rp[j] - mx);
  red[threadIdx.x] = sum; __syncthreads();
  for (int s = 128; s > 0; s >>= 1) {
    if (threadIdx.x < s) red[threadIdx.x] += red[threadIdx.x + s];
    __syncthreads();
  }
  const float inv = 1.0f / red[0];
  for (int j = threadIdx.x; j < cols; j += 256) rp[j] = __expf(rp[j] - mx) * inv;
}

// s[m, 0:256]=emb[x[m]], s[m,256:768]=cfeat[m], s[m,768:1280]=m_out[m]; bf16 out
__global__ __launch_bounds__(256)
void build_s(const int* __restrict__ x, const float* __restrict__ emb,
             const float* __restrict__ cfeat, const float* __restrict__ mout,
             unsigned short* __restrict__ s)
{
  const size_t idx = (size_t)blockIdx.x * 256 + threadIdx.x;
  if (idx >= (size_t)1024 * 1280) return;
  const int m = (int)(idx / 1280);
  const int j = (int)(idx - (size_t)m * 1280);
  float val;
  if (j < 256)      val = emb  [(size_t)x[m] * 256 + j];
  else if (j < 768) val = cfeat[(size_t)m * 512 + (j - 256)];
  else              val = mout [(size_t)m * 512 + (j - 768)];
  s[idx] = f2bf(val);
}

// ---------------------------------------------------------------------------
// Orchestration
// ---------------------------------------------------------------------------
static inline unsigned nblk(size_t n) { return (unsigned)((n + 255) / 256); }

extern "C" void kernel_launch(void* const* d_in, const int* in_sizes, int n_in,
                              void* d_out, int out_size, void* d_ws, size_t ws_size,
                              hipStream_t stream)
{
  const int*   x    = (const int*)  d_in[0];   // [32,32]
  const int*   c    = (const int*)  d_in[1];   // [32,32,64]
  const float* emb  = (const float*)d_in[2];   // [32000,256]
  const float *w[5], *bb[5], *g[5], *be[5];
  for (int i = 0; i < 5; ++i) {
    w [i] = (const float*)d_in[3 + 4 * i + 0];
    bb[i] = (const float*)d_in[3 + 4 * i + 1];
    g [i] = (const float*)d_in[3 + 4 * i + 2];
    be[i] = (const float*)d_in[3 + 4 * i + 3];
  }
  const float* fcW  = (const float*)d_in[23];  // [1280,32000]
  const float* fcb  = (const float*)d_in[24];  // [32000]
  const float* memK = (const float*)d_in[25];  // [4096,512]
  const float* memV = (const float*)d_in[26];  // [4096,512]
  float* out = (float*)d_out;                  // [32,32,32000]

  const int MR = 1024;                         // B*S rows

  // bump allocator over d_ws
  char* wsb = (char*)d_ws; size_t off = 0;
  auto alloc = [&](size_t bytes) -> void* {
    off = (off + 255) & ~(size_t)255;
    void* p = wsb + off; off += bytes; return p;
  };

  float* ce = (float*)alloc((size_t)MR * 64 * 256 * 4);   // conv input [1024,64,256]
  float* h1 = (float*)alloc((size_t)MR * 32 * 127 * 4);
  float* h2 = (float*)alloc((size_t)MR * 64 * 62  * 4);
  float* h3 = (float*)alloc((size_t)MR * 128 * 30 * 4);
  float* h4 = (float*)alloc((size_t)MR * 256 * 14 * 4);
  unsigned short* h4b   = (unsigned short*)alloc((size_t)MR * 3584 * 2);
  unsigned short* w5b   = (unsigned short*)alloc((size_t)512 * 3584 * 2);
  float* h5             = (float*)alloc((size_t)MR * 512 * 4);     // cfeat f32
  unsigned short* cfb   = (unsigned short*)alloc((size_t)MR * 512 * 2);
  unsigned short* mKb   = (unsigned short*)alloc((size_t)4096 * 512 * 2);
  unsigned short* mVtb  = (unsigned short*)alloc((size_t)512 * 4096 * 2);
  float* scores         = (float*)alloc((size_t)MR * 4096 * 4);
  unsigned short* attnb = (unsigned short*)alloc((size_t)MR * 4096 * 2);
  float* mout           = (float*)alloc((size_t)MR * 512 * 4);
  unsigned short* sb    = (unsigned short*)alloc((size_t)MR * 1280 * 2);
  unsigned short* fcWtb = (unsigned short*)alloc((size_t)32000 * 1280 * 2);
  float* mu = (float*)alloc(512 * 4);
  float* rs = (float*)alloc(512 * 4);

  // 0) weight prep (independent of activations)
  cvt_bf16          <<<nblk((size_t)512 * 3584),   256, 0, stream>>>(w[4], w5b, (size_t)512 * 3584);
  cvt_bf16          <<<nblk((size_t)4096 * 512),   256, 0, stream>>>(memK, mKb, (size_t)4096 * 512);
  transpose_cvt_bf16<<<nblk((size_t)4096 * 512),   256, 0, stream>>>(memV, mVtb, 4096, 512);
  transpose_cvt_bf16<<<nblk((size_t)1280 * 32000), 256, 0, stream>>>(fcW, fcWtb, 1280, 32000);

  // 1) context embedding gather -> conv input
  gather_embed<<<nblk((size_t)MR * 64 * 256), 256, 0, stream>>>(c, emb, ce, MR * 64);

  // 2) conv stack (conv+bias+leaky, then BN with batch stats)
  conv1d_leaky<<<nblk((size_t)MR * 32 * 127), 256, 0, stream>>>(ce, w[0], bb[0], h1, MR, 64, 256, 32, 127, 8, 2, 2);
  bn_stats<<<32, 256, 0, stream>>>(h1, mu, rs, MR, 32, 127);
  bn_norm <<<nblk((size_t)MR * 32 * 127), 256, 0, stream>>>(h1, mu, rs, g[0], be[0], MR, 32, 127);

  conv1d_leaky<<<nblk((size_t)MR * 64 * 62), 256, 0, stream>>>(h1, w[1], bb[1], h2, MR, 32, 127, 64, 62, 8, 2, 2);
  bn_stats<<<64, 256, 0, stream>>>(h2, mu, rs, MR, 64, 62);
  bn_norm <<<nblk((size_t)MR * 64 * 62), 256, 0, stream>>>(h2, mu, rs, g[1], be[1], MR, 64, 62);

  conv1d_leaky<<<nblk((size_t)MR * 128 * 30), 256, 0, stream>>>(h2, w[2], bb[2], h3, MR, 64, 62, 128, 30, 8, 2, 2);
  bn_stats<<<128, 256, 0, stream>>>(h3, mu, rs, MR, 128, 30);
  bn_norm <<<nblk((size_t)MR * 128 * 30), 256, 0, stream>>>(h3, mu, rs, g[2], be[2], MR, 128, 30);

  conv1d_leaky<<<nblk((size_t)MR * 256 * 14), 256, 0, stream>>>(h3, w[3], bb[3], h4, MR, 128, 30, 256, 14, 8, 2, 2);
  bn_stats<<<256, 256, 0, stream>>>(h4, mu, rs, MR, 256, 14);
  bn_norm <<<nblk((size_t)MR * 256 * 14), 256, 0, stream>>>(h4, mu, rs, g[3], be[3], MR, 256, 14);

  // 3) conv5 (L_out=1) as WMMA GEMM: [1024,3584] @ [3584,512], fused bias+leaky
  cvt_bf16<<<nblk((size_t)MR * 3584), 256, 0, stream>>>(h4, h4b, (size_t)MR * 3584);
  gemm_bf16_wmma<<<dim3(512 / 128, MR / 64), 256, 0, stream>>>(h4b, w5b, h5, MR, 512, 3584, bb[4], 1.0f, 1);
  bn_stats<<<512, 256, 0, stream>>>(h5, mu, rs, MR, 512, 1);
  bn_norm <<<nblk((size_t)MR * 512), 256, 0, stream>>>(h5, mu, rs, g[4], be[4], MR, 512, 1);

  // 4) memory attention: scores = cfeat @ memK^T / sqrt(512); softmax; @ memV
  cvt_bf16<<<nblk((size_t)MR * 512), 256, 0, stream>>>(h5, cfb, (size_t)MR * 512);
  gemm_bf16_wmma<<<dim3(4096 / 128, MR / 64), 256, 0, stream>>>(cfb, mKb, scores, MR, 4096, 512,
                                                                nullptr, 0.04419417382415922f, 0);
  softmax4096_bf16<<<MR, 256, 0, stream>>>(scores, attnb);
  gemm_bf16_wmma<<<dim3(512 / 128, MR / 64), 256, 0, stream>>>(attnb, mVtb, mout, MR, 512, 4096,
                                                               nullptr, 1.0f, 0);

  // 5) concat -> FC GEMM (dominant: 1024 x 32000 x 1280) -> softmax
  build_s<<<nblk((size_t)MR * 1280), 256, 0, stream>>>(x, emb, h5, mout, sb);
  gemm_bf16_wmma<<<dim3(32000 / 128, MR / 64), 256, 0, stream>>>(sb, fcWtb, out, MR, 32000, 1280,
                                                                 fcb, 1.0f, 0);
  softmax_inplace<<<MR, 256, 0, stream>>>(out, 32000);
}
